// InfoNCELoss_42863773614157
// MI455X (gfx1250) — matrix operand order repaired and checked by reading, not compiled, not run on hardware
//
#include <hip/hip_runtime.h>

// ---------------------------------------------------------------------------
// InfoNCE (NT-Xent) loss, fused for MI455X / gfx1250.
//
//   z = rownorm(concat(proj_1, proj_2))        [8192 x 1024]  (f16 in ws)
//   sim = z z^T  (never materialized)          137 GFLOP, f16 WMMA, f32 acc
//   denom[r] = sum_{c != r} exp(2*sim[r,c])
//   pos[r]   = sim[r, r ^ 4096]
//   loss = mean(log(denom) - 2*pos)
//
// rep(f16) = 16 MB -> resident in the 192 MB L2 after first touch.
// B panels (64 cols x 256 K = 32 KB) are DMA'd into LDS by the Tensor Data
// Mover (double-buffered, TENSORcnt-tracked) and shared by the 8 waves of a
// workgroup; A rows (16 x 1024 = 32 KB/wave) stay hot in the WGP cache.
// ---------------------------------------------------------------------------

typedef __attribute__((ext_vector_type(16))) _Float16 v16h;
typedef __attribute__((ext_vector_type(8)))  _Float16 v8h;
typedef __attribute__((ext_vector_type(8)))  float    v8f;
typedef __attribute__((ext_vector_type(4)))  unsigned u32x4;
typedef __attribute__((ext_vector_type(8)))  int      i32x8;
typedef __attribute__((ext_vector_type(4)))  int      i32x4;

#define BHALF   4096          // B (rows per view)
#define DDIM    1024          // feature dim
#define NROWS   8192          // 2B
#define INV_T   2.0f          // 1 / temperature (T = 0.5)

#define NCHUNK  2048          // sim columns handled per workgroup (grid.y=4)
#define NB      64            // B-panel columns per stage
#define KB      256           // B-panel K depth per stage (panel = 32 KB)
#define NT      4             // 16-wide N-tiles per wave (NB / 16)
#define KSTAGES (DDIM / KB)   // 4 K stages per column group
#define NGROUPS (NCHUNK / NB) // 32 column groups per block
#define NSTAGES (NGROUPS * KSTAGES)

#if __has_builtin(__builtin_amdgcn_tensor_load_to_lds) && \
    __has_builtin(__builtin_amdgcn_s_wait_tensorcnt)
#define HAVE_TDM 1
#else
#define HAVE_TDM 0
#endif

// ---------------------------------------------------------------------------
// Kernel 1: L2-normalize each row, emit f16 rep, zero denom.
// ---------------------------------------------------------------------------
__global__ __launch_bounds__(256)
void nce_normalize(const float* __restrict__ p1, const float* __restrict__ p2,
                   _Float16* __restrict__ rep, float* __restrict__ denom) {
    const int r   = blockIdx.x;
    const int tid = threadIdx.x;
    const float* src = (r < BHALF) ? (p1 + (size_t)r * DDIM)
                                   : (p2 + (size_t)(r - BHALF) * DDIM);
    float v[4];
    float ss = 0.f;
#pragma unroll
    for (int i = 0; i < 4; ++i) {
        v[i] = src[tid + i * 256];
        ss += v[i] * v[i];
    }
    __shared__ float red[256];
    red[tid] = ss;
    __syncthreads();
    for (int s = 128; s > 0; s >>= 1) {
        if (tid < s) red[tid] += red[tid + s];
        __syncthreads();
    }
    const float inv = rsqrtf(red[0]);
    _Float16* dst = rep + (size_t)r * DDIM;
#pragma unroll
    for (int i = 0; i < 4; ++i)
        dst[tid + i * 256] = (_Float16)(v[i] * inv);
    if (tid == 0) denom[r] = 0.f;
}

// ---------------------------------------------------------------------------
// Kernel 2: fused sim = rep rep^T with online exp/row-sum/positive capture.
// grid = (NROWS/128, NROWS/NCHUNK), block = 256 (8 waves).
//   wave w owns rows [blockIdx.x*128 + w*16, +16)
//   block owns cols [blockIdx.y*NCHUNK, +NCHUNK)
// ---------------------------------------------------------------------------
__global__ __launch_bounds__(256)
void nce_sim_fused(const _Float16* __restrict__ rep,
                   float* __restrict__ denom, float* __restrict__ pos) {
    const int lane = threadIdx.x & 31;
    const int wave = threadIdx.x >> 5;
    const int n    = lane & 15;      // column-within-tile (A: row-within-tile)
    const int half = lane >> 4;      // lane-half selects K group / M group

    const int Rm  = blockIdx.x * 128 + wave * 16;  // 16-row M tile base
    const int Nb0 = blockIdx.y * NCHUNK;           // column range base

    // A-matrix rows: lanes 0-15 and 16-31 both carry M = lane&15 (ISA layout).
    const _Float16* __restrict__ arow = rep + (Rm + n) * DDIM;

    __shared__ _Float16 Bs[2][NB * KB];            // 2 x 32 KB double buffer

#if HAVE_TDM
    // Issue one Tensor-Data-Mover DMA: panel for stage s -> Bs[b].
    // D# per ISA 8.3/8.4: 2-D tensor (1024 x 8192 f16), 256x64 tile.
    // This toolchain exposes the 6-operand builtin: descriptor groups 0..3
    // (2-D tensor -> groups 2/3 zero/disabled) + trailing group + cpol.
    auto tdm_issue = [&](int s, int b) {
        const int ng = s / KSTAGES;
        const int ks = (s % KSTAGES) * KB;
        unsigned long long ga =
            (unsigned long long)(uintptr_t)(rep + (Nb0 + ng * NB) * DDIM + ks);
        unsigned lds = (unsigned)(uintptr_t)(&Bs[b][0]);   // low 32 = LDS offset
        u32x4 g0 = { 1u,                                   // count=1, user mode
                     lds,                                  // lds_addr
                     (unsigned)ga,                         // global_addr[31:0]
                     (unsigned)((ga >> 32) & 0x1FFFFFFu) | (2u << 30) }; // +type=2
        i32x8 g1 = { 0x10000,                  // data_size=1 (2 B), no multicast
                     (int)((unsigned)DDIM << 16),   // tensor_dim0 = 1024
                     (int)((unsigned)NROWS << 16),  // tensor_dim1 = 8192
                     (int)((unsigned)KB << 16),     // tile_dim0   = 256 (K)
                     NB,                            // tile_dim1   = 64 cols
                     DDIM,                          // tensor_dim0_stride = 1024
                     0, 0 };
        i32x4 gz4 = { 0, 0, 0, 0 };
        i32x8 gz8 = { 0, 0, 0, 0, 0, 0, 0, 0 };
        __builtin_amdgcn_tensor_load_to_lds(g0, g1, gz4, gz4, gz8, 0);
    };
    if (wave == 0) tdm_issue(0, 0);
#endif

    float rsum[8];
#pragma unroll
    for (int i = 0; i < 8; ++i) rsum[i] = 0.f;

    const v8f vzero = {0.f, 0.f, 0.f, 0.f, 0.f, 0.f, 0.f, 0.f};

    for (int ng = 0; ng < NGROUPS; ++ng) {
        // Real (once-per-group) accumulator reset: keeps the hot stage loop
        // free of per-stage cndmask chains on the 32 acc VGPRs.
        v8f acc[NT];
#pragma unroll
        for (int t = 0; t < NT; ++t) acc[t] = vzero;

        for (int ksi = 0; ksi < KSTAGES; ++ksi) {
            const int s   = ng * KSTAGES + ksi;
            const int buf = s & 1;
            const int ks  = ksi * KB;

#if HAVE_TDM
            // Pipeline: wait panel s, publish, start DMA of panel s+1 into
            // the other buffer (all waves finished reading it at the barrier).
            if (wave == 0) __builtin_amdgcn_s_wait_tensorcnt((short)0);
            __syncthreads();
            if (wave == 0 && s + 1 < NSTAGES) tdm_issue(s + 1, buf ^ 1);
#else
            // Fallback: cooperative copy, 128 B per thread.
            __syncthreads();
            {
                const int col  = threadIdx.x >> 2;
                const int koff = (threadIdx.x & 3) * 64;
                const _Float16* g =
                    rep + (Nb0 + ng * NB + col) * DDIM + ks + koff;
                _Float16* l = &Bs[buf][col * KB + koff];
#pragma unroll
                for (int j = 0; j < 8; ++j)
                    *(v8h*)(l + j * 8) = *(const v8h*)(g + j * 8);
            }
            __syncthreads();
#endif

            // ---- 8 WMMA K-steps over the staged panel --------------------
#pragma unroll
            for (int kb = 0; kb < KB; kb += 32) {
                // A fragment, 16x32 f16 (ISA 7.12.2): lanes 0-15 carry
                // K = {kb..kb+7, kb+16..kb+23}; lanes 16-31 shifted by 8.
                const _Float16* ap = arow + ks + kb + 8 * half;
                v8h alo = *(const v8h*)(ap);
                v8h ahi = *(const v8h*)(ap + 16);
                v16h a = __builtin_shufflevector(alo, ahi,
                        0,1,2,3,4,5,6,7,8,9,10,11,12,13,14,15);

                // Preload all NT B fragments into distinct registers so the
                // ds_loads clause together and the WMMAs issue back-to-back.
                v16h bfrag[NT];
#pragma unroll
                for (int t = 0; t < NT; ++t) {
                    // B fragment, 32x16 f16: lanes 0-15 K={kb..kb+7, kb+16..kb+23};
                    // lanes 16-31 shifted by 8 (matches A).
                    const _Float16* bp =
                        &Bs[buf][(t * 16 + n) * KB + kb + 8 * half];
                    v8h blo = *(const v8h*)(bp);
                    v8h bhi = *(const v8h*)(bp + 16);
                    bfrag[t] = __builtin_shufflevector(blo, bhi,
                            0,1,2,3,4,5,6,7,8,9,10,11,12,13,14,15);
                }
#pragma unroll
                for (int t = 0; t < NT; ++t)
                    acc[t] = __builtin_amdgcn_wmma_f32_16x16x32_f16(
                            false, a, false, bfrag[t], (short)0, acc[t],
                            false, false);
            }
        }

        // ---- fused epilogue for this 64-column group ---------------------
        // C/D layout: lane n = lane&15; VGPR v holds row m = v + 8*half.
#pragma unroll
        for (int t = 0; t < NT; ++t) {
            const int c = Nb0 + ng * NB + t * 16 + n;       // global column
#pragma unroll
            for (int v = 0; v < 8; ++v) {
                const int r = Rm + v + 8 * half;            // global row
                const float sv = acc[t][v];
                float e = __expf(sv * INV_T);
                if (c == r) e = 0.f;                        // mask self-sim
                if (c == (r ^ BHALF)) pos[r] = sv;          // unique writer
                rsum[v] += e;
            }
        }
    }

    // Reduce row sums across the 16 lanes sharing a row (bits 0..3 of lane),
    // then one f32 atomic per row per column-chunk (NROWS * grid.y total).
#pragma unroll
    for (int v = 0; v < 8; ++v) {
        float x = rsum[v];
        x += __shfl_xor(x, 1, 32);
        x += __shfl_xor(x, 2, 32);
        x += __shfl_xor(x, 4, 32);
        x += __shfl_xor(x, 8, 32);
        if (n == 0)
            atomicAdd(&denom[Rm + v + 8 * half], x);
    }
}

// ---------------------------------------------------------------------------
// Kernel 3: loss = mean(log(denom) - 2*pos)
// ---------------------------------------------------------------------------
__global__ __launch_bounds__(256)
void nce_loss(const float* __restrict__ denom, const float* __restrict__ pos,
              float* __restrict__ out) {
    const int tid = threadIdx.x;
    float acc = 0.f;
    for (int r = tid; r < NROWS; r += 256)
        acc += logf(denom[r]) - pos[r] * INV_T;
    __shared__ float red[256];
    red[tid] = acc;
    __syncthreads();
    for (int s = 128; s > 0; s >>= 1) {
        if (tid < s) red[tid] += red[tid + s];
        __syncthreads();
    }
    if (tid == 0) out[0] = red[0] / (float)NROWS;
}

// ---------------------------------------------------------------------------
extern "C" void kernel_launch(void* const* d_in, const int* in_sizes, int n_in,
                              void* d_out, int out_size, void* d_ws, size_t ws_size,
                              hipStream_t stream) {
    const float* p1 = (const float*)d_in[0];
    const float* p2 = (const float*)d_in[1];

    // ws layout: rep f16 [8192*1024] (16 MB) | denom f32 [8192] | pos f32 [8192]
    _Float16* rep   = (_Float16*)d_ws;
    float*    denom = (float*)((char*)d_ws + (size_t)NROWS * DDIM * sizeof(_Float16));
    float*    pos   = denom + NROWS;
    float*    out   = (float*)d_out;

    nce_normalize<<<NROWS, 256, 0, stream>>>(p1, p2, rep, denom);

    dim3 grid(NROWS / 128, NROWS / NCHUNK);        // (64, 4)
    nce_sim_fused<<<grid, 256, 0, stream>>>(rep, denom, pos);

    nce_loss<<<1, 256, 0, stream>>>(denom, pos, out);
}
